// CrystalGraphNetwork_65420941853354
// MI455X (gfx1250) — matrix operand order repaired and checked by reading, not compiled, not run on hardware
//
#include <hip/hip_runtime.h>
#include <hip/hip_bf16.h>

typedef __bf16 bf16;
typedef __attribute__((ext_vector_type(16))) __bf16 v16bf;
typedef __attribute__((ext_vector_type(8)))  float  v8f;
typedef __attribute__((ext_vector_type(4)))  unsigned int u32x4;

#define N_NODES  50000
#define N_EDGES  1600000
#define N_GRAPHS 512
#define ATOM_DIM 128
#define EDGE_DIM 41
#define HIDDEN   256
#define KPAD     192      // 128 (x_j) + 41 (edge_f) padded to 6 k-tiles of 32
#define TILE_E   64       // edges per workgroup tile (4 M-tiles of 16)

// ---------- helpers ----------
__device__ __forceinline__ bf16 f2bf(float f) {
    unsigned u = __builtin_bit_cast(unsigned, f);
    unsigned r = u + 0x7fffu + ((u >> 16) & 1u);      // round-to-nearest-even
    unsigned short h = (unsigned short)(r >> 16);
    return __builtin_bit_cast(bf16, h);
}
// silu via v_rcp_f32 instead of IEEE divide (avoids div_scale/div_fmas chains)
__device__ __forceinline__ float silu(float v) {
    return v * __builtin_amdgcn_rcpf(1.0f + __expf(-v));
}

// A fragment (16x32 bf16, row-major tile in LDS, row stride `stride` halves)
// lanes 0-15 : row=lane,    K-chunks [0..7] and [16..23]
// lanes 16-31: row=lane-16, K-chunks [8..15] and [24..31]
__device__ __forceinline__ v16bf frag_a(const bf16* tile, int stride, int kt, int lane) {
    int r  = lane & 15;
    int kb = kt * 32 + ((lane < 16) ? 0 : 8);
    const bf16* p = tile + r * stride + kb;
    union { v16bf v; u32x4 u[2]; } f;
    f.u[0] = *reinterpret_cast<const u32x4*>(p);
    f.u[1] = *reinterpret_cast<const u32x4*>(p + 16);
    return f.v;
}
// B fragment (32x16 bf16) from N-major (transposed) weight cache in LDS:
// lane n holds K-chunks like A: lanes 0-15 K=[0..7],[16..23]; lanes 16-31 K=[8..15],[24..31]
__device__ __forceinline__ v16bf frag_b(const bf16* tileT, int stride, int kt, int lane) {
    int n  = lane & 15;
    int kb = kt * 32 + ((lane < 16) ? 0 : 8);
    const bf16* p = tileT + n * stride + kb;
    union { v16bf v; u32x4 u[2]; } f;
    f.u[0] = *reinterpret_cast<const u32x4*>(p);
    f.u[1] = *reinterpret_cast<const u32x4*>(p + 16);
    return f.v;
}

// ---------- kernels ----------
__global__ void k_zero(float* buf, size_t n) {
    size_t i = (size_t)blockIdx.x * blockDim.x + threadIdx.x;
    size_t stride = (size_t)gridDim.x * blockDim.x;
    for (; i < n; i += stride) buf[i] = 0.0f;
}

__global__ void k_embed(const int* Z, const float* emb, float* x) {
    int i = blockIdx.x * blockDim.x + threadIdx.x;
    if (i >= N_NODES * ATOM_DIM) return;
    int n = i >> 7, c = i & 127;
    x[i] = emb[Z[n] * ATOM_DIM + c];
}

// Fused edge MLP: m = silu(silu([x_j|e_f] @ W1 + b1) @ W2 + b2); agg[edge_i] += m
__global__ void __launch_bounds__(256)
k_edge(const int* __restrict__ edge_i, const int* __restrict__ edge_j,
       const float* __restrict__ edge_f, const float* __restrict__ x,
       const float* __restrict__ w1, const float* __restrict__ b1,
       const float* __restrict__ w2, const float* __restrict__ b2,
       float* __restrict__ agg, int layer) {
    __shared__ __align__(16) bf16 sW1T[HIDDEN][KPAD];    //  98304 B (N-major)
    __shared__ __align__(16) bf16 sW2T[HIDDEN][HIDDEN];  // 131072 B (N-major)
    __shared__ __align__(16) bf16 sA[TILE_E][KPAD];      //  24576 B
    __shared__ __align__(16) bf16 sH[TILE_E][HIDDEN];    //  32768 B
    __shared__ float sB1[HIDDEN];
    __shared__ float sB2[HIDDEN];
    __shared__ int   sEI[TILE_E];
    __shared__ int   sEJ[TILE_E];

    const int tid  = threadIdx.x;
    const int wave = tid >> 5;
    const int lane = tid & 31;

    // cache weights (transposed -> N-major) + biases in LDS, once per block
    const float* W1 = w1 + (size_t)layer * (ATOM_DIM + EDGE_DIM) * HIDDEN;
    for (int idx = tid; idx < HIDDEN * KPAD; idx += blockDim.x) {
        int n = idx / KPAD, k = idx % KPAD;
        float v = (k < ATOM_DIM + EDGE_DIM) ? W1[k * HIDDEN + n] : 0.0f;
        sW1T[n][k] = f2bf(v);
    }
    const float* W2 = w2 + (size_t)layer * HIDDEN * HIDDEN;
    for (int idx = tid; idx < HIDDEN * HIDDEN; idx += blockDim.x) {
        int n = idx >> 8, k = idx & 255;
        sW2T[n][k] = f2bf(W2[k * HIDDEN + n]);
    }
    for (int i = tid; i < HIDDEN; i += blockDim.x) {
        sB1[i] = b1[layer * HIDDEN + i];
        sB2[i] = b2[layer * HIDDEN + i];
    }
    __syncthreads();

    const int ntiles = N_EDGES / TILE_E;   // 25000, exact
    for (int t = blockIdx.x; t < ntiles; t += gridDim.x) {
        const int e0 = t * TILE_E;
        if (tid < TILE_E) {
            sEI[tid] = edge_i[e0 + tid];
            sEJ[tid] = edge_j[e0 + tid];
        }
        __syncthreads();
        // build bf16 A tile: [x_j | edge_f | 0-pad]
        for (int idx = tid; idx < TILE_E * KPAD; idx += blockDim.x) {
            int m = idx / KPAD, k = idx % KPAD;
            float v;
            if (k < ATOM_DIM)                 v = x[(size_t)sEJ[m] * ATOM_DIM + k];
            else if (k < ATOM_DIM + EDGE_DIM) v = edge_f[(size_t)(e0 + m) * EDGE_DIM + (k - ATOM_DIM)];
            else                              v = 0.0f;
            sA[m][k] = f2bf(v);
        }
        __syncthreads();

        // GEMM1: (64 x 192) @ (192 x 256) -> silu -> sH (bf16)
        // mt = p & 3 is invariant under p += 8 -> each wave stays on one M-tile
        for (int p = wave; p < 4 * 16; p += 8) {
            int mt = p & 3, nt = p >> 2;
            v8f acc = {};
            #pragma unroll
            for (int kt = 0; kt < KPAD / 32; ++kt) {
                v16bf a = frag_a(&sA[mt * 16][0], KPAD, kt, lane);
                v16bf b = frag_b(&sW1T[nt * 16][0], KPAD, kt, lane);
                acc = __builtin_amdgcn_wmma_f32_16x16x32_bf16(false, a, false, b,
                                                              (short)0, acc, false, false);
            }
            int col   = lane & 15;
            int rbase = mt * 16 + ((lane < 16) ? 0 : 8);
            float bias = sB1[nt * 16 + col];
            #pragma unroll
            for (int v = 0; v < 8; ++v) {
                float h = silu(acc[v] + bias);
                sH[rbase + v][nt * 16 + col] = f2bf(h);
            }
        }
        __syncthreads();

        // GEMM2: (64 x 256) @ (256 x 256) -> silu -> atomic scatter into agg
        for (int p = wave; p < 4 * 16; p += 8) {
            int mt = p & 3, nt = p >> 2;
            v8f acc = {};
            #pragma unroll
            for (int kt = 0; kt < HIDDEN / 32; ++kt) {
                v16bf a = frag_a(&sH[mt * 16][0], HIDDEN, kt, lane);
                v16bf b = frag_b(&sW2T[nt * 16][0], HIDDEN, kt, lane);
                acc = __builtin_amdgcn_wmma_f32_16x16x32_bf16(false, a, false, b,
                                                              (short)0, acc, false, false);
            }
            int col   = lane & 15;
            int rbase = mt * 16 + ((lane < 16) ? 0 : 8);
            float bias = sB2[nt * 16 + col];
            #pragma unroll
            for (int v = 0; v < 8; ++v) {
                float m = silu(acc[v] + bias);
                int node = sEI[rbase + v];
                atomicAdd(&agg[(size_t)node * HIDDEN + nt * 16 + col], m);
            }
        }
        __syncthreads();
    }
}

// x = LN(silu(x + agg @ wlin + blin)) ; one block (128 thr) per node
__global__ void __launch_bounds__(128)
k_node(const float* __restrict__ agg, const float* __restrict__ wlin,
       const float* __restrict__ blin, const float* __restrict__ ln_g,
       const float* __restrict__ ln_b, float* __restrict__ x, int layer) {
    int n = blockIdx.x;
    int c = threadIdx.x;
    __shared__ float sAgg[HIDDEN];
    __shared__ float red[ATOM_DIM];
    for (int k = c; k < HIDDEN; k += ATOM_DIM) sAgg[k] = agg[(size_t)n * HIDDEN + k];
    __syncthreads();
    const float* W = wlin + (size_t)layer * HIDDEN * ATOM_DIM;
    float acc = blin[layer * ATOM_DIM + c];
    #pragma unroll 4
    for (int k = 0; k < HIDDEN; ++k) acc += sAgg[k] * W[k * ATOM_DIM + c];
    float xv = silu(x[(size_t)n * ATOM_DIM + c] + acc);
    red[c] = xv; __syncthreads();
    for (int s = 64; s > 0; s >>= 1) { if (c < s) red[c] += red[c + s]; __syncthreads(); }
    float mu = red[0] * (1.0f / ATOM_DIM);
    __syncthreads();
    float d = xv - mu;
    red[c] = d * d; __syncthreads();
    for (int s = 64; s > 0; s >>= 1) { if (c < s) red[c] += red[c + s]; __syncthreads(); }
    float var = red[0] * (1.0f / ATOM_DIM);
    float r = rsqrtf(var + 1e-5f);
    x[(size_t)n * ATOM_DIM + c] = d * r * ln_g[layer * ATOM_DIM + c] + ln_b[layer * ATOM_DIM + c];
}

__global__ void k_pool(const float* __restrict__ x, const int* __restrict__ batch,
                       float* __restrict__ sums, float* __restrict__ counts) {
    int i = blockIdx.x * blockDim.x + threadIdx.x;
    if (i >= N_NODES * ATOM_DIM) return;
    int n = i >> 7, c = i & 127;
    int g = batch[n];
    atomicAdd(&sums[g * ATOM_DIM + c], x[i]);
    if (c == 0) atomicAdd(&counts[g], 1.0f);
}

__global__ void __launch_bounds__(256)
k_head(const float* __restrict__ sums, const float* __restrict__ counts,
       const float* __restrict__ hw1, const float* __restrict__ hb1,
       const float* __restrict__ hw2, const float* __restrict__ hb2,
       const float* __restrict__ hw3, const float* __restrict__ hb3,
       float* __restrict__ out) {
    int g = blockIdx.x;
    int t = threadIdx.x;
    __shared__ float sp[ATOM_DIM];
    __shared__ float sh1[HIDDEN];
    __shared__ float red[ATOM_DIM];
    float cnt = fmaxf(counts[g], 1.0f);
    if (t < ATOM_DIM) sp[t] = sums[g * ATOM_DIM + t] * __builtin_amdgcn_rcpf(cnt);
    __syncthreads();
    float a = hb1[t];
    for (int k = 0; k < ATOM_DIM; ++k) a += sp[k] * hw1[k * HIDDEN + t];
    sh1[t] = silu(a);
    __syncthreads();
    if (t < ATOM_DIM) {
        float b = hb2[t];
        for (int k = 0; k < HIDDEN; ++k) b += sh1[k] * hw2[k * ATOM_DIM + t];
        red[t] = silu(b) * hw3[t];
    }
    __syncthreads();
    for (int s = 64; s > 0; s >>= 1) { if (t < s) red[t] += red[t + s]; __syncthreads(); }
    if (t == 0) out[g] = red[0] + hb3[0];
}

// ---------- launcher ----------
extern "C" void kernel_launch(void* const* d_in, const int* in_sizes, int n_in,
                              void* d_out, int out_size, void* d_ws, size_t ws_size,
                              hipStream_t stream) {
    const int*   Z      = (const int*)  d_in[0];
    const int*   edge_i = (const int*)  d_in[1];
    const int*   edge_j = (const int*)  d_in[2];
    const float* edge_f = (const float*)d_in[3];
    const int*   batch  = (const int*)  d_in[4];
    const float* emb    = (const float*)d_in[5];
    const float* w1     = (const float*)d_in[6];
    const float* b1     = (const float*)d_in[7];
    const float* w2     = (const float*)d_in[8];
    const float* b2     = (const float*)d_in[9];
    const float* wlin   = (const float*)d_in[10];
    const float* blin   = (const float*)d_in[11];
    const float* ln_g   = (const float*)d_in[12];
    const float* ln_b   = (const float*)d_in[13];
    const float* hw1    = (const float*)d_in[14];
    const float* hb1    = (const float*)d_in[15];
    const float* hw2    = (const float*)d_in[16];
    const float* hb2    = (const float*)d_in[17];
    const float* hw3    = (const float*)d_in[18];
    const float* hb3    = (const float*)d_in[19];
    float* out = (float*)d_out;

    float* xbuf   = (float*)d_ws;                          // 50000*128
    float* agg    = xbuf + (size_t)N_NODES * ATOM_DIM;     // 50000*256
    float* sums   = agg  + (size_t)N_NODES * HIDDEN;       // 512*128
    float* counts = sums + (size_t)N_GRAPHS * ATOM_DIM;    // 512

    k_embed<<<(N_NODES * ATOM_DIM + 255) / 256, 256, 0, stream>>>(Z, emb, xbuf);

    for (int l = 0; l < 3; ++l) {
        k_zero<<<2048, 256, 0, stream>>>(agg, (size_t)N_NODES * HIDDEN);
        k_edge<<<1024, 256, 0, stream>>>(edge_i, edge_j, edge_f, xbuf,
                                         w1, b1, w2, b2, agg, l);
        k_node<<<N_NODES, 128, 0, stream>>>(agg, wlin, blin, ln_g, ln_b, xbuf, l);
    }

    k_zero<<<64, 256, 0, stream>>>(sums, (size_t)(N_GRAPHS * ATOM_DIM + N_GRAPHS));
    k_pool<<<(N_NODES * ATOM_DIM + 255) / 256, 256, 0, stream>>>(xbuf, batch, sums, counts);
    k_head<<<N_GRAPHS, 256, 0, stream>>>(sums, counts, hw1, hb1, hw2, hb2, hw3, hb3, out);
}
